// HierarchicalMemoryBank_37477884625218
// MI455X (gfx1250) — matrix operand order repaired and checked by reading, not compiled, not run on hardware
//
#include <hip/hip_runtime.h>

// ---------------- problem constants ----------------
#define BB 8
#define LL 4096
#define DD 1024
#define NN (BB * LL)          // 32768 rows
#define NG_ 1024
#define NL_ 256
#define NC_ (NG_ + NL_)       // 1280 combined codes
#define EPSN 1e-12f
#define INV_TEMP (1.0f / 0.07f)

typedef __attribute__((ext_vector_type(16))) __bf16 v16bf;
typedef __attribute__((ext_vector_type(8)))  float  v8f;

// ---------------- WMMA wrapper ----------------
__device__ __forceinline__ v8f wmma_bf16(v16bf a, v16bf b, v8f c) {
  return __builtin_amdgcn_wmma_f32_16x16x32_bf16(false, a, false, b, (short)0, c, false, false);
}

// ---------------- CDNA5 async global->LDS copy (ASYNCcnt path) ----------------
__device__ __forceinline__ unsigned lds_off_of(const void* p) {
  // LDS aperture: addr[31:0] is the wave-relative LDS byte address (ISA 10.2)
  return (unsigned)(unsigned long long)p;
}
__device__ __forceinline__ void async_b128(unsigned lds_off, const void* g) {
  asm volatile("global_load_async_to_lds_b128 %0, %1, off"
               :: "v"(lds_off), "v"((unsigned long long)g)
               : "memory");
}
__device__ __forceinline__ void wait_async0() {
  asm volatile("s_wait_asynccnt 0x0" ::: "memory");
}
__device__ __forceinline__ void wait_async2() {
  asm volatile("s_wait_asynccnt 0x2" ::: "memory");
}

// A fragment: 16x32 bf16, row-major source with leading dim ld.
// lane&15 = M row; lane>>4 selects K-halves {0..7,16..23} vs {8..15,24..31}.
__device__ __forceinline__ v16bf load_a_frag(const __bf16* p, int ld) {
  int lane = threadIdx.x & 31;
  const __bf16* row = p + (size_t)(lane & 15) * ld;
  int half = lane >> 4;
  v16bf r;
#pragma unroll
  for (int v = 0; v < 8; ++v) {
    int k = (v < 4) ? (2 * v + 8 * half) : (16 + 2 * (v - 4) + 8 * half);
    r[2 * v]     = row[k];
    r[2 * v + 1] = row[k + 1];
  }
  return r;
}

// B fragment from an [Ncol][K]-major buffer (i.e. B already transposed):
// lane&15 = N col; lane>>4 selects K-halves {0..7,16..23} vs {8..15,24..31}.
__device__ __forceinline__ v16bf load_bT_frag(const __bf16* p, int ld) {
  int lane = threadIdx.x & 31;
  const __bf16* row = p + (size_t)(lane & 15) * ld;
  int kh = lane >> 4;
  v16bf r;
#pragma unroll
  for (int v = 0; v < 8; ++v) {
    int k = (v < 4) ? (2 * v + 8 * kh) : (16 + 2 * (v - 4) + 8 * kh);
    r[2 * v]     = row[k];
    r[2 * v + 1] = row[k + 1];
  }
  return r;
}

// ---------------- block reduce (256 threads) ----------------
__device__ __forceinline__ float blockReduceSum(float v) {
  __shared__ float sh[8];
  __shared__ float bc;
  int lane = threadIdx.x & 31, w = threadIdx.x >> 5;
#pragma unroll
  for (int off = 16; off; off >>= 1) v += __shfl_xor(v, off, 32);
  __syncthreads();
  if (lane == 0) sh[w] = v;
  __syncthreads();
  float s = (threadIdx.x < 8) ? sh[threadIdx.x] : 0.f;
  if (w == 0) {
#pragma unroll
    for (int off = 4; off; off >>= 1) s += __shfl_xor(s, off, 32);
  }
  if (threadIdx.x == 0) bc = s;
  __syncthreads();
  return bc;
}

// ---------------- small utility kernels ----------------
__global__ void zero_f32(float* p, int n) {
  int i = blockIdx.x * 256 + threadIdx.x;
  if (i < n) p[i] = 0.f;
}

__global__ void cvt_f32_bf16(const float* __restrict__ s, __bf16* __restrict__ d, int n) {
  int i = blockIdx.x * 256 + threadIdx.x;
  if (i < n) d[i] = (__bf16)s[i];
}

// per-row: z -> bf16; u -> copy-out + l2norm -> bf16
__global__ void prep_rows(const float* __restrict__ z, const float* __restrict__ u,
                          __bf16* __restrict__ zb, __bf16* __restrict__ ubn,
                          float* __restrict__ u_out) {
  size_t row = blockIdx.x;
  int t = threadIdx.x;
  const float* zr = z + row * DD;
  const float* ur = u + row * DD;
  float uv[4]; float ss = 0.f;
#pragma unroll
  for (int j = 0; j < 4; ++j) {
    int c = t + 256 * j;
    float zv = zr[c];
    zb[row * DD + c] = (__bf16)zv;
    float x = ur[c];
    uv[j] = x;
    u_out[row * DD + c] = x;
    ss += x * x;
  }
  ss = blockReduceSum(ss);
  float inv = 1.f / fmaxf(sqrtf(ss), EPSN);
#pragma unroll
  for (int j = 0; j < 4; ++j) {
    int c = t + 256 * j;
    ubn[row * DD + c] = (__bf16)(uv[j] * inv);
  }
}

// z_norm = l2norm(z_proj) -> bf16 ; loss_direct partial = |dot(z_norm, u_norm)|
__global__ void znorm_dot(const float* __restrict__ zp, const __bf16* __restrict__ ubn,
                          __bf16* __restrict__ znb, float* __restrict__ accum) {
  size_t row = blockIdx.x;
  int t = threadIdx.x;
  float zv[4]; float ss = 0.f;
#pragma unroll
  for (int j = 0; j < 4; ++j) {
    int c = t + 256 * j;
    float x = zp[row * DD + c];
    zv[j] = x; ss += x * x;
  }
  ss = blockReduceSum(ss);
  float inv = 1.f / fmaxf(sqrtf(ss), EPSN);
  float dot = 0.f;
#pragma unroll
  for (int j = 0; j < 4; ++j) {
    int c = t + 256 * j;
    float zn = zv[j] * inv;
    znb[row * DD + c] = (__bf16)zn;
    dot += zn * (float)ubn[row * DD + c];
  }
  dot = blockReduceSum(dot);
  if (t == 0) atomicAdd(&accum[2], fabsf(dot));
}

// ---------------- dict sim + argmax + scatter-add (WMMA) ----------------
template <int NCODES>
__global__ __launch_bounds__(32) void sim_argmax_scatter(
    const __bf16* __restrict__ un,     // [N, D] normalized noise, bf16
    const __bf16* __restrict__ dictb,  // [NCODES, D] bf16 (acts as B^T)
    float* __restrict__ sums,          // [NCODES, D]
    float* __restrict__ counts) {      // [NCODES]
  __shared__ int bestIdx[16];
  int lane = threadIdx.x;
  size_t rowBase = (size_t)blockIdx.x * 16;
  float bv[8]; int bi[8];
#pragma unroll
  for (int v = 0; v < 8; ++v) { bv[v] = -3.0e38f; bi[v] = 0; }

  for (int ct = 0; ct < NCODES / 16; ++ct) {
    v8f acc = {0.f, 0.f, 0.f, 0.f, 0.f, 0.f, 0.f, 0.f};
    for (int ks = 0; ks < DD; ks += 32) {
      v16bf a = load_a_frag(un + rowBase * DD + ks, DD);
      v16bf b = load_bT_frag(dictb + (size_t)ct * 16 * DD + ks, DD);
      acc = wmma_bf16(a, b, acc);
    }
    int col = ct * 16 + (lane & 15);
#pragma unroll
    for (int v = 0; v < 8; ++v) {
      float s = acc[v];
      if (s > bv[v]) { bv[v] = s; bi[v] = col; }
    }
  }
#pragma unroll
  for (int v = 0; v < 8; ++v) {
#pragma unroll
    for (int off = 8; off; off >>= 1) {
      float ov = __shfl_xor(bv[v], off, 16);
      int   oi = __shfl_xor(bi[v], off, 16);
      if (ov > bv[v] || (ov == bv[v] && oi < bi[v])) { bv[v] = ov; bi[v] = oi; }
    }
  }
  if (lane == 0) {
#pragma unroll
    for (int v = 0; v < 8; ++v) bestIdx[v] = bi[v];
  }
  if (lane == 16) {
#pragma unroll
    for (int v = 0; v < 8; ++v) bestIdx[8 + v] = bi[v];
  }
  __syncthreads();
  for (int r = 0; r < 16; ++r) {
    int idx = bestIdx[r];
#pragma unroll
    for (int j = 0; j < DD / 32; ++j) {
      int c = lane + 32 * j;
      atomicAdd(&sums[(size_t)idx * DD + c], (float)un[(rowBase + r) * DD + c]);
    }
    if (lane == 0) atomicAdd(&counts[idx], 1.0f);
  }
}

// ---------------- dict finalize: EMA + l2norm -> combined (row + transposed) ----
__global__ void dict_finalize(const float* __restrict__ gdict, const float* __restrict__ ldict,
                              const float* __restrict__ sums, const float* __restrict__ counts,
                              __bf16* __restrict__ comb, __bf16* __restrict__ combT) {
  int i = blockIdx.x;
  int t = threadIdx.x;
  const float* dold = (i < NG_) ? (gdict + (size_t)i * DD) : (ldict + (size_t)(i - NG_) * DD);
  float m = (i < NG_) ? 0.999f : 0.8f;
  float cnt = counts[i];
  float nv[4]; float ss = 0.f;
#pragma unroll
  for (int j = 0; j < 4; ++j) {
    int c = t + 256 * j;
    float od = dold[c];
    float x = (cnt > 0.f) ? (m * od + (1.f - m) * (sums[(size_t)i * DD + c] / fmaxf(cnt, 1.f)))
                          : od;
    nv[j] = x; ss += x * x;
  }
  ss = blockReduceSum(ss);
  float inv = 1.f / fmaxf(sqrtf(ss), EPSN);
#pragma unroll
  for (int j = 0; j < 4; ++j) {
    int c = t + 256 * j;
    __bf16 b = (__bf16)(nv[j] * inv);
    comb[(size_t)i * DD + c] = b;
    combT[(size_t)c * NC_ + i] = b;
  }
}

// ---------------- tiled bf16 WMMA GEMM, async double-buffered A ----------------
// C[M,Ncol] = act(A[M,K] @ B[K,Ncol] + bias)
// EPI 0: relu -> bf16 | 1: none -> f32 + bf16 | 2: sigmoid -> bf16
template <int EPI>
__global__ __launch_bounds__(256) void gemm_bf16k(
    const __bf16* __restrict__ A, const __bf16* __restrict__ B,
    const float* __restrict__ bias, float* __restrict__ Cf,
    __bf16* __restrict__ Cb, int M, int K, int Ncol) {
  constexpr int BM = 128, BN = 64, BK = 32;
  __shared__ alignas(16) __bf16 As[2][BM * BK];
  __shared__ alignas(16) __bf16 Bs[2][BN * BK];   // transposed: [n][k]
  int tid = threadIdx.x, lane = tid & 31, wave = tid >> 5;
  int wr = wave & 3, wc = wave >> 2;
  size_t rowBase = (size_t)blockIdx.x * BM;
  size_t colBase = (size_t)blockIdx.y * BN;
  v8f zero = {0.f, 0.f, 0.f, 0.f, 0.f, 0.f, 0.f, 0.f};
  v8f acc[2][2] = {{zero, zero}, {zero, zero}};

  // per-thread staging coordinates
  int aflat = tid * 16;
  int ar = aflat >> 5, ac = aflat & 31;      // A: 16 bf16 per thread
  int bflat = tid * 8;
  int bk = bflat >> 6, bn = bflat & 63;      // B: 8 bf16 per thread

  // ---- async stage of A tile kt into LDS buffer ----
  auto stageA = [&](int kt, __bf16* dst) {
    const __bf16* src = A + (rowBase + ar) * (size_t)K + kt + ac;
    unsigned l0 = lds_off_of(dst + ar * BK + ac);
    async_b128(l0, src);
    async_b128(l0 + 16, src + 8);            // +8 bf16 = +16 bytes
  };
  // ---- regular stage of B tile kt (transpose into [n][k]) ----
  auto stageB = [&](int kt, __bf16* dst) {
    const float4* src = (const float4*)(B + (size_t)(kt + bk) * Ncol + colBase + bn);
    float4 v = src[0];
    alignas(16) __bf16 tmp[8];
    *(float4*)tmp = v;
#pragma unroll
    for (int q = 0; q < 8; ++q) dst[(bn + q) * BK + bk] = tmp[q];
  };

  stageA(0, As[0]);
  stageB(0, Bs[0]);
  int buf = 0;
  for (int kt = 0; kt < K; kt += BK) {
    int nxt = buf ^ 1;
    if (kt + BK < K) {
      stageA(kt + BK, As[nxt]);              // prefetch next A tile (async)
      stageB(kt + BK, Bs[nxt]);
      __builtin_prefetch(B + (size_t)(kt + 2 * BK + bk) * Ncol + colBase + bn, 0, 1);
      wait_async2();                          // current tile's 2 asyncs retired
    } else {
      wait_async0();
    }
    __syncthreads();
#pragma unroll
    for (int i = 0; i < 2; ++i) {
      v16bf a = load_a_frag(&As[buf][(wr * 32 + i * 16) * BK], BK);
#pragma unroll
      for (int j = 0; j < 2; ++j) {
        v16bf b = load_bT_frag(&Bs[buf][(wc * 32 + j * 16) * BK], BK);
        acc[i][j] = wmma_bf16(a, b, acc[i][j]);
      }
    }
    __syncthreads();
    buf = nxt;
  }

  int n0 = lane & 15, half = lane >> 4;
#pragma unroll
  for (int i = 0; i < 2; ++i) {
#pragma unroll
    for (int j = 0; j < 2; ++j) {
      size_t gc = colBase + wc * 32 + j * 16 + n0;
      float bv = bias[gc];
#pragma unroll
      for (int v = 0; v < 8; ++v) {
        size_t gr = rowBase + wr * 32 + i * 16 + v + 8 * half;
        float x = acc[i][j][v] + bv;
        size_t o = gr * Ncol + gc;
        if constexpr (EPI == 0) { x = fmaxf(x, 0.f); Cb[o] = (__bf16)x; }
        else if constexpr (EPI == 1) { Cf[o] = x; Cb[o] = (__bf16)x; }
        else { x = 1.f / (1.f + __expf(-x)); Cb[o] = (__bf16)x; }
      }
    }
  }
}

// ---------------- fused: S = z_norm@combined^T, losses, softmax,
//                  weighted_u = softmax@combined, z_clean = z - gate*weighted_u
__global__ __launch_bounds__(32) void softmax_weighted(
    const __bf16* __restrict__ znb, const __bf16* __restrict__ comb,
    const __bf16* __restrict__ combT, const __bf16* __restrict__ gateb,
    const float* __restrict__ zflat, float* __restrict__ out,
    float* __restrict__ accum) {
  __shared__ alignas(16) __bf16 Az[16 * DD];   // 32 KB: cached z_norm row block
  __shared__ __bf16 S[16 * NC_];               // 40 KB: score/weight tile
  __shared__ float rowmax[16], rowsum[16];
  int lane = threadIdx.x;
  size_t rowBase = (size_t)blockIdx.x * 16;
  int n0 = lane & 15, half = lane >> 4;

  // async-load the 16xD z_norm block into LDS once (read 80x in phase 1)
  for (int it = 0; it < 64; ++it) {
    int flat = (it * 32 + lane) * 8;           // 8 bf16 (16 B) per lane-issue
    int r = flat >> 10, c = flat & 1023;
    async_b128(lds_off_of(&Az[r * DD + c]), znb + (rowBase + r) * (size_t)DD + c);
  }
  wait_async0();
  __syncthreads();

  float lmax[8];
#pragma unroll
  for (int v = 0; v < 8; ++v) lmax[v] = -3.0e38f;
  float lg = 0.f, llo = 0.f;

  // phase 1: score tiles over 1280 codes (K = 1024), A from LDS
  for (int ct = 0; ct < NC_ / 16; ++ct) {
    v8f acc = {0.f, 0.f, 0.f, 0.f, 0.f, 0.f, 0.f, 0.f};
    for (int ks = 0; ks < DD; ks += 32) {
      v16bf a = load_a_frag(&Az[ks], DD);
      v16bf b = load_bT_frag(comb + (size_t)ct * 16 * DD + ks, DD);
      acc = wmma_bf16(a, b, acc);
    }
    int cg = ct * 16 + n0;
#pragma unroll
    for (int v = 0; v < 8; ++v) {
      float s = acc[v];
      if (cg < NG_) lg += fabsf(s); else llo += fabsf(s);
      lmax[v] = fmaxf(lmax[v], s);
      int r = v + 8 * half;
      S[r * NC_ + cg] = (__bf16)s;
    }
  }
#pragma unroll
  for (int v = 0; v < 8; ++v) {
#pragma unroll
    for (int off = 8; off; off >>= 1) lmax[v] = fmaxf(lmax[v], __shfl_xor(lmax[v], off, 16));
  }
  if (lane == 0) {
#pragma unroll
    for (int v = 0; v < 8; ++v) rowmax[v] = lmax[v];
  }
  if (lane == 16) {
#pragma unroll
    for (int v = 0; v < 8; ++v) rowmax[8 + v] = lmax[v];
  }
#pragma unroll
  for (int off = 16; off; off >>= 1) { lg += __shfl_xor(lg, off, 32); llo += __shfl_xor(llo, off, 32); }
  if (lane == 0) { atomicAdd(&accum[0], lg); atomicAdd(&accum[1], llo); }
  __syncthreads();

  // phase 2: in-place exp, row sums (2 lanes per row)
  {
    int r = lane >> 1, c0 = lane & 1;
    float rmax = rowmax[r];
    float rs = 0.f;
    for (int j = 0; j < NC_ / 2; ++j) {
      int c = c0 + 2 * j;
      float s = (float)S[r * NC_ + c];
      float e = __expf((s - rmax) * INV_TEMP);
      S[r * NC_ + c] = (__bf16)e;
      rs += e;
    }
    rs += __shfl_xor(rs, 1, 32);
    if (c0 == 0) rowsum[r] = rs;
  }
  __syncthreads();

  // phase 3: weighted_u = expS @ combined (K = 1280), fused gate epilogue
  for (int ct2 = 0; ct2 < DD / 16; ++ct2) {
    v8f acc = {0.f, 0.f, 0.f, 0.f, 0.f, 0.f, 0.f, 0.f};
    for (int ks = 0; ks < NC_; ks += 32) {
      v16bf a = load_a_frag(&S[ks], NC_);
      v16bf b = load_bT_frag(combT + (size_t)ct2 * 16 * NC_ + ks, NC_);
      acc = wmma_bf16(a, b, acc);
    }
#pragma unroll
    for (int v = 0; v < 8; ++v) {
      int r = v + 8 * half;
      size_t gr = rowBase + r;
      size_t gc = (size_t)ct2 * 16 + n0;
      float wu = acc[v] / rowsum[r];
      float g = (float)gateb[gr * DD + gc];
      out[gr * DD + gc] = zflat[gr * DD + gc] - g * wu;
    }
  }
}

__global__ void loss_final(const float* __restrict__ accum, float* __restrict__ o) {
  o[0] = accum[0] * (1.f / ((float)NN * (float)NG_)) +
         accum[1] * (1.f / ((float)NN * (float)NL_)) +
         accum[2] * (1.f / (float)NN);
}

// ---------------- host launcher ----------------
extern "C" void kernel_launch(void* const* d_in, const int* in_sizes, int n_in,
                              void* d_out, int out_size, void* d_ws, size_t ws_size,
                              hipStream_t stream) {
  const float* z     = (const float*)d_in[0];
  const float* u     = (const float*)d_in[1];
  const float* W1    = (const float*)d_in[2];
  const float* b1    = (const float*)d_in[3];
  const float* W2    = (const float*)d_in[4];
  const float* b2    = (const float*)d_in[5];
  const float* Wg    = (const float*)d_in[6];
  const float* bg    = (const float*)d_in[7];
  const float* gdict = (const float*)d_in[8];
  const float* ldict = (const float*)d_in[9];

  float* out        = (float*)d_out;
  float* out_zclean = out;
  float* out_loss   = out + (size_t)NN * DD;
  float* out_zproj  = out + (size_t)NN * DD + 1;
  float* out_uflat  = out + 2 * (size_t)NN * DD + 1;

  // workspace layout (bf16 then f32; reuse: zb->zprojb, hb->znormb, ub->gateb)
  __bf16* zb    = (__bf16*)d_ws;
  __bf16* hb    = zb  + (size_t)NN * DD;
  __bf16* ub    = hb  + (size_t)NN * DD;
  __bf16* w1b   = ub  + (size_t)NN * DD;
  __bf16* w2b   = w1b + (size_t)DD * DD;
  __bf16* wgb   = w2b + (size_t)DD * DD;
  __bf16* gdb   = wgb + (size_t)DD * DD;
  __bf16* ldb   = gdb + (size_t)NG_ * DD;
  __bf16* comb  = ldb + (size_t)NL_ * DD;
  __bf16* combT = comb + (size_t)NC_ * DD;
  float*  sums   = (float*)(combT + (size_t)DD * NC_);
  float*  counts = sums + (size_t)NC_ * DD;
  float*  accum  = counts + NC_;

  const int DD2 = DD * DD;

  // 0) zero dict accumulators + loss accumulators (contiguous)
  {
    int n = NC_ * DD + NC_ + 4;
    zero_f32<<<(n + 255) / 256, 256, 0, stream>>>(sums, n);
  }
  // 1) convert weights & dicts to bf16
  cvt_f32_bf16<<<(DD2 + 255) / 256, 256, 0, stream>>>(W1, w1b, DD2);
  cvt_f32_bf16<<<(DD2 + 255) / 256, 256, 0, stream>>>(W2, w2b, DD2);
  cvt_f32_bf16<<<(DD2 + 255) / 256, 256, 0, stream>>>(Wg, wgb, DD2);
  cvt_f32_bf16<<<(NG_ * DD + 255) / 256, 256, 0, stream>>>(gdict, gdb, NG_ * DD);
  cvt_f32_bf16<<<(NL_ * DD + 255) / 256, 256, 0, stream>>>(ldict, ldb, NL_ * DD);
  // 2) per-row prep: z->bf16, u->copy-out + l2norm->bf16
  prep_rows<<<NN, 256, 0, stream>>>(z, u, zb, ub, out_uflat);
  // 3) dict sim+argmax+scatter (WMMA), then EMA+norm finalize -> combined
  sim_argmax_scatter<NG_><<<NN / 16, 32, 0, stream>>>(ub, gdb, sums, counts);
  sim_argmax_scatter<NL_><<<NN / 16, 32, 0, stream>>>(ub, ldb, sums + (size_t)NG_ * DD, counts + NG_);
  dict_finalize<<<NC_, 256, 0, stream>>>(gdict, ldict, sums, counts, comb, combT);
  // 4) projector GEMMs (async double-buffered A staging)
  dim3 gg(NN / 128, DD / 64);
  gemm_bf16k<0><<<gg, 256, 0, stream>>>(zb, w1b, b1, nullptr, hb, NN, DD, DD);            // h = relu(zW1+b1)
  gemm_bf16k<1><<<gg, 256, 0, stream>>>(hb, w2b, b2, out_zproj, zb, NN, DD, DD);          // z_proj (f32 out + bf16)
  // 5) z_norm + loss_direct (needs ub before it becomes gate)
  znorm_dot<<<NN, 256, 0, stream>>>(out_zproj, ub, hb, accum);                            // hb := z_norm bf16
  // 6) gate = sigmoid(z_proj Wg + bg) -> ub
  gemm_bf16k<2><<<gg, 256, 0, stream>>>(zb, wgb, bg, nullptr, ub, NN, DD, DD);
  // 7) fused scores/softmax/weighted combine/gated subtract
  softmax_weighted<<<NN / 16, 32, 0, stream>>>(hb, comb, combT, ub, z, out_zclean, accum);
  // 8) scalar loss
  loss_final<<<1, 1, 0, stream>>>(accum, out_loss);
}
